// KGCN_42039139893748
// MI455X (gfx1250) — compile-verified, matrix operand order checked by reading, not run on hardware
//
#include <hip/hip_runtime.h>
#include <math.h>

typedef __attribute__((ext_vector_type(16))) _Float16 v16h;
typedef __attribute__((ext_vector_type(8)))  float    v8f;

#define DIM   64
#define NNB   16
#define WAVES 4   // waves per block (128 threads)

__device__ __forceinline__ float sigmoidf_(float x) { return 1.0f / (1.0f + __expf(-x)); }

// K offset inside a 32-wide chunk for packed f16 fragment pair j (0..7), half h (0..1)
// ISA 7.12.2: 16-bit A 16x32: VGPR0..3 -> K {0..7 | 8..15}, VGPR4..7 -> K {16..23 | 24..31}
__device__ __forceinline__ int kbase_(int j, int h) {
    return (j < 4) ? (2 * j + 8 * h) : (16 + 2 * (j - 4) + 8 * h);
}

// 16x64 tile (rows in LDS, f32) times W^T (f16 in LDS) + bias, activation, write back to tile.
// ACT: 0 = sigmoid, 1 = tanh
template <int ACT>
__device__ void tile_proj_wmma(float (*x)[DIM], const _Float16 (*Wh)[DIM],
                               const float* __restrict__ bias, int lane) {
    const int h = lane >> 4, ln = lane & 15;

    // A fragments: row M = ln, two K chunks
    v16h A0, A1;
    {
        const float* xr = x[ln];
#pragma unroll
        for (int j = 0; j < 8; ++j) {
            const int kb = kbase_(j, h);
            A0[2 * j]     = (_Float16)xr[kb];
            A0[2 * j + 1] = (_Float16)xr[kb + 1];
            A1[2 * j]     = (_Float16)xr[32 + kb];
            A1[2 * j + 1] = (_Float16)xr[32 + kb + 1];
        }
    }
#pragma unroll
    for (int n = 0; n < 4; ++n) {
        const int n0 = n * 16;
        // B fragments: B[k][ncol] = W[n0+ncol][k] ; lane holds column N = ln
        v16h B0, B1;
        const _Float16* wc = Wh[n0 + ln];
#pragma unroll
        for (int j = 0; j < 8; ++j) {
            const int kb = kbase_(j, h);
            B0[2 * j]     = wc[kb];
            B0[2 * j + 1] = wc[kb + 1];
            B1[2 * j]     = wc[32 + kb];
            B1[2 * j + 1] = wc[32 + kb + 1];
        }
        v8f c;
        const float bb = bias[n0 + ln];
#pragma unroll
        for (int i = 0; i < 8; ++i) c[i] = bb;
        c = __builtin_amdgcn_wmma_f32_16x16x32_f16(false, A0, false, B0, (short)0, c, false, false);
        c = __builtin_amdgcn_wmma_f32_16x16x32_f16(false, A1, false, B1, (short)0, c, false, false);
#pragma unroll
        for (int i = 0; i < 8; ++i) {
            const int row = i + 8 * h;            // D layout: M = i + 8*(lane>=16)
            float r = (ACT == 0) ? sigmoidf_(c[i]) : tanhf(c[i]);
            x[row][n0 + ln] = r;                  // A frags already in regs; safe to overwrite
        }
    }
}

// ---------------- Kernel 1: hop-1 nodes (B*16 rows) -> H1 (f16) ----------------
__global__ __launch_bounds__(128) void kgcn_hop1(
    const int* __restrict__ u, const int* __restrict__ v,
    const int* __restrict__ adj_ent, const int* __restrict__ adj_rel,
    const float* __restrict__ E, const float* __restrict__ R,
    const float* __restrict__ W, const float* __restrict__ bias,
    _Float16* __restrict__ H1)
{
    __shared__ _Float16 Wh[DIM][DIM];                 // 8 KB, shared by block
    __shared__ float s_user[WAVES][DIM];
    __shared__ float s_alpha[WAVES][16][NNB];
    __shared__ int   s_ent2[WAVES][16][NNB];
    __shared__ int   s_e1[WAVES][16];
    __shared__ float s_x[WAVES][16][DIM];

    const int tid = threadIdx.x;
    for (int i = tid; i < DIM * DIM; i += blockDim.x)
        ((_Float16*)Wh)[i] = (_Float16)W[i];

    const int wave = tid >> 5, lane = tid & 31;
    const int h = lane >> 4, ln = lane & 15;
    const int b = blockIdx.x * WAVES + wave;          // batch element

    { // user embedding (2 floats per lane)
        const int uidx = u[b];
        const float2 uu = ((const float2*)(E + (size_t)uidx * DIM))[lane];
        s_user[wave][2 * lane]     = uu.x;
        s_user[wave][2 * lane + 1] = uu.y;
    }
    if (lane < 16) s_e1[wave][lane] = adj_ent[(size_t)v[b] * NNB + lane];
    __syncthreads();

    // --- attention scores + softmax: 2 rows per pass (one per half-wave) ---
    for (int jp = 0; jp < 8; ++jp) {
        const int row = 2 * jp + h;
        const int e1 = s_e1[wave][row];
        const int r  = adj_rel[(size_t)e1 * NNB + ln];
        s_ent2[wave][row][ln] = adj_ent[(size_t)e1 * NNB + ln];
        const float* rp = R + (size_t)r * DIM;
        const float* up = s_user[wave];
        float sc = 0.0f;
        for (int d = 0; d < DIM; d += 4)
            sc += up[d] * rp[d] + up[d + 1] * rp[d + 1] +
                  up[d + 2] * rp[d + 2] + up[d + 3] * rp[d + 3];
        float mx = sc;
        for (int m = 8; m >= 1; m >>= 1) mx = fmaxf(mx, __shfl_xor(mx, m, 16));
        const float e = __expf(sc - mx);
        float sm = e;
        for (int m = 8; m >= 1; m >>= 1) sm += __shfl_xor(sm, m, 16);
        s_alpha[wave][row][ln] = e / sm;
    }
    __syncthreads();

    // --- x = e1 + sum_k alpha_k * E[ent2_k]  (lane owns dims 2*lane, 2*lane+1) ---
    for (int row = 0; row < 16; ++row) {
        const int e1 = s_e1[wave][row];
        const float2 base = ((const float2*)(E + (size_t)e1 * DIM))[lane];
        float ax = base.x, ay = base.y;
        for (int k = 0; k < NNB; ++k) {
            const float a  = s_alpha[wave][row][k];
            const int   ei = s_ent2[wave][row][k];
            const float2 ee = ((const float2*)(E + (size_t)ei * DIM))[lane];
            ax = fmaf(a, ee.x, ax);
            ay = fmaf(a, ee.y, ay);
        }
        s_x[wave][row][2 * lane]     = ax;
        s_x[wave][row][2 * lane + 1] = ay;
    }
    __syncthreads();

    // --- projection + sigmoid via WMMA, result into s_x ---
    tile_proj_wmma<0>(s_x[wave], Wh, bias, lane);
    __syncthreads();

    // --- store H1 as f16 (halves stage-2 gather traffic) ---
    for (int row = 0; row < 16; ++row) {
        const float2 xv = *(const float2*)&s_x[wave][row][2 * lane];
        _Float16* hp = H1 + ((size_t)b * NNB + row) * DIM + 2 * lane;
        hp[0] = (_Float16)xv.x;
        hp[1] = (_Float16)xv.y;
    }
}

// ---------------- Kernel 2: hop-0 both iterations + readout ----------------
__global__ __launch_bounds__(128) void kgcn_final(
    const int* __restrict__ u, const int* __restrict__ v,
    const int* __restrict__ adj_ent, const int* __restrict__ adj_rel,
    const float* __restrict__ E, const float* __restrict__ R,
    const float* __restrict__ W, const float* __restrict__ bias,
    const _Float16* __restrict__ H1, float* __restrict__ out)
{
    __shared__ _Float16 Wh[DIM][DIM];                 // 8 KB
    __shared__ float s_user[WAVES][16][DIM];          // 16 KB
    __shared__ float s_alpha[WAVES][16][NNB];
    __shared__ int   s_ent1[WAVES][16][NNB];
    __shared__ int   s_v[WAVES][16];
    __shared__ float s_x[WAVES][16][DIM];             // 16 KB

    const int tid = threadIdx.x;
    for (int i = tid; i < DIM * DIM; i += blockDim.x)
        ((_Float16*)Wh)[i] = (_Float16)W[i];

    const int wave = tid >> 5, lane = tid & 31;
    const int h = lane >> 4, ln = lane & 15;
    const int b0 = (blockIdx.x * WAVES + wave) * 16;  // 16 batch elements per wave

    if (lane < 16) s_v[wave][lane] = v[b0 + lane];
    for (int j = 0; j < 16; ++j) {
        const int uidx = u[b0 + j];
        const float2 uu = ((const float2*)(E + (size_t)uidx * DIM))[lane];
        s_user[wave][j][2 * lane]     = uu.x;
        s_user[wave][j][2 * lane + 1] = uu.y;
    }
    __syncthreads();

    // --- alpha0: shared by iteration 0 and iteration 1 (scores only depend on user & rel) ---
    for (int jp = 0; jp < 8; ++jp) {
        const int row = 2 * jp + h;
        const int vb = s_v[wave][row];
        const int r  = adj_rel[(size_t)vb * NNB + ln];
        s_ent1[wave][row][ln] = adj_ent[(size_t)vb * NNB + ln];
        const float* rp = R + (size_t)r * DIM;
        const float* up = s_user[wave][row];
        float sc = 0.0f;
        for (int d = 0; d < DIM; d += 4)
            sc += up[d] * rp[d] + up[d + 1] * rp[d + 1] +
                  up[d + 2] * rp[d + 2] + up[d + 3] * rp[d + 3];
        float mx = sc;
        for (int m = 8; m >= 1; m >>= 1) mx = fmaxf(mx, __shfl_xor(mx, m, 16));
        const float e = __expf(sc - mx);
        float sm = e;
        for (int m = 8; m >= 1; m >>= 1) sm += __shfl_xor(sm, m, 16);
        s_alpha[wave][row][ln] = e / sm;
    }
    __syncthreads();

    // --- iteration 0, hop 0: x_a = E[v] + sum alpha * E[ent1] ---
    for (int row = 0; row < 16; ++row) {
        const int vb = s_v[wave][row];
        const float2 base = ((const float2*)(E + (size_t)vb * DIM))[lane];
        float ax = base.x, ay = base.y;
        for (int k = 0; k < NNB; ++k) {
            const float a  = s_alpha[wave][row][k];
            const int   ei = s_ent1[wave][row][k];
            const float2 ee = ((const float2*)(E + (size_t)ei * DIM))[lane];
            ax = fmaf(a, ee.x, ax);
            ay = fmaf(a, ee.y, ay);
        }
        s_x[wave][row][2 * lane]     = ax;
        s_x[wave][row][2 * lane + 1] = ay;
    }
    __syncthreads();

    tile_proj_wmma<0>(s_x[wave], Wh, bias, lane);     // h0 = sigmoid(x_a W^T + b)
    __syncthreads();

    // --- iteration 1, hop 0: x_b = h0 + sum alpha * H1[b][k] ---
    for (int row = 0; row < 16; ++row) {
        float ax = s_x[wave][row][2 * lane];
        float ay = s_x[wave][row][2 * lane + 1];
        for (int k = 0; k < NNB; ++k) {
            const float a = s_alpha[wave][row][k];
            const _Float16* hp = H1 + ((size_t)(b0 + row) * NNB + k) * DIM + 2 * lane;
            ax = fmaf(a, (float)hp[0], ax);
            ay = fmaf(a, (float)hp[1], ay);
        }
        s_x[wave][row][2 * lane]     = ax;
        s_x[wave][row][2 * lane + 1] = ay;
    }
    __syncthreads();

    tile_proj_wmma<1>(s_x[wave], Wh, bias, lane);     // item = tanh(x_b W^T + b)
    __syncthreads();

    // --- readout: sigmoid(<user, item>) ---
    if (lane < 16) {
        const float* up = s_user[wave][lane];
        const float* ip = s_x[wave][lane];
        float d = 0.0f;
        for (int dd = 0; dd < DIM; ++dd) d = fmaf(up[dd], ip[dd], d);
        out[b0 + lane] = sigmoidf_(d);
    }
}

extern "C" void kernel_launch(void* const* d_in, const int* in_sizes, int n_in,
                              void* d_out, int out_size, void* d_ws, size_t ws_size,
                              hipStream_t stream) {
    const int*   u       = (const int*)d_in[0];
    const int*   v       = (const int*)d_in[1];
    const int*   adj_ent = (const int*)d_in[2];
    const int*   adj_rel = (const int*)d_in[3];
    const float* E       = (const float*)d_in[4];
    const float* R       = (const float*)d_in[5];
    const float* W       = (const float*)d_in[6];
    const float* bias    = (const float*)d_in[7];
    float*       out     = (float*)d_out;
    _Float16*    H1      = (_Float16*)d_ws;           // B*16*64 f16 = 8 MB

    const int B = in_sizes[0];                        // 4096
    const int blocksA = B / WAVES;                    // 1 wave per batch element
    const int blocksB = B / (WAVES * 16);             // 1 wave per 16 batch elements

    kgcn_hop1 <<<blocksA, 128, 0, stream>>>(u, v, adj_ent, adj_rel, E, R, W, bias, H1);
    kgcn_final<<<blocksB, 128, 0, stream>>>(u, v, adj_ent, adj_rel, E, R, W, bias, H1, out);
}
